// GraphEmbedder_64372969832919
// MI455X (gfx1250) — matrix operand order repaired and checked by reading, not compiled, not run on hardware
//
#include <hip/hip_runtime.h>
#include <hip/hip_bf16.h>
#include <cmath>
#include <math.h>

// ---------------------------------------------------------------------------
// GraphEmbedder forward for MI455X (gfx1250, wave32).
// Dense node-feature GEMMs use v_wmma_f32_16x16x32_f16 (f16 in, f32 acc),
// with the shared B tile staged into LDS via global_load_async_to_lds_b128
// (ASYNCcnt) and read back with ds_load_b128.
// Edge attention / segment softmax / pooling use fp32 global atomics.
// ---------------------------------------------------------------------------

typedef __attribute__((ext_vector_type(16))) _Float16 v16h;
typedef __attribute__((ext_vector_type(8)))  _Float16 v8h;
typedef __attribute__((ext_vector_type(8)))  float    v8f;

__device__ __forceinline__ float gelu_f(float x) {
    return 0.5f * x * (1.0f + erff(x * 0.70710678118654752f));
}

// float atomic max via signed/unsigned int ordering trick
__device__ __forceinline__ void atomicMaxF(float* a, float v) {
    if (v >= 0.0f) atomicMax((int*)a, __float_as_int(v));
    else           atomicMin((unsigned int*)a, __float_as_uint(v));
}

// ------------------------------- elementwise -------------------------------

__global__ void k_fill(float* __restrict__ p, float v, long n) {
    long t = blockIdx.x * (long)blockDim.x + threadIdx.x;
    if (t < n) p[t] = v;
}

__global__ void k_h16(const float* __restrict__ in, _Float16* __restrict__ out, long n) {
    long t = blockIdx.x * (long)blockDim.x + threadIdx.x;
    if (t < n) out[t] = (_Float16)in[t];
}

// W[K,M] row-major f32 -> Wt[M,K] f16 (column-major of W)
__global__ void k_w16t(const float* __restrict__ W, _Float16* __restrict__ Wt, int K, int M) {
    long t = blockIdx.x * (long)blockDim.x + threadIdx.x;
    if (t >= (long)K * M) return;
    int k = (int)(t / M), m = (int)(t % M);
    Wt[(long)m * K + k] = (_Float16)W[t];
}

__global__ void k_add_inplace(float* __restrict__ dst, const float* __restrict__ src, long n) {
    long t = blockIdx.x * (long)blockDim.x + threadIdx.x;
    if (t < n) dst[t] += src[t];
}

__global__ void k_copy(float* __restrict__ dst, const float* __restrict__ src, long n) {
    long t = blockIdx.x * (long)blockDim.x + threadIdx.x;
    if (t < n) dst[t] = src[t];
}

// ------------------------------ WMMA GEMM ----------------------------------
// C[N,M] = A[N,K](f16) * W[K,M] + bias ;  Bt is W pre-transposed: Bt[m*K+k].
// Block = 128 threads = 4 wave32s. Block tile = 64 rows x 64 cols.
// Each wave computes a 16x64 strip: one A fragment reused across 4 WMMAs.
// The 64x32 B tile (4 KB) is shared by all 4 waves: staged into LDS with
// global_load_async_to_lds_b128 once per K step, then read back per-wave
// with ds_load_b128. The LDS destination address is taken from the *actual*
// address of sB (flat-address low 32 bits == LDS byte address per the
// aperture rules), which also makes sB escape into the asm so the compiler
// keeps the LDS array and the ds reads alive.

__global__ void __launch_bounds__(128)
k_wmma_gemm_bias(const _Float16* __restrict__ A, const _Float16* __restrict__ Bt,
                 const float* __restrict__ bias, float* __restrict__ C,
                 int N, int K, int M)
{
    // LDS B tile, column-major: col*32 + k  (halves); 64 bytes per column.
    __shared__ _Float16 sB[64 * 32];

    const int tid  = threadIdx.x;
    const int lane = tid & 31;
    const int wave = tid >> 5;
    const int row0  = (blockIdx.y << 6) + (wave << 4);  // 16-row strip per wave
    const int colB0 = blockIdx.x << 6;                  // 64-col block tile
    const int mr    = lane & 15;   // M row (A) / N col (B) within tile
    const int kh    = lane >> 4;   // K half-group select

    const int  ar     = row0 + mr;
    const bool avalid = ar < N;
    const _Float16* arow = A + (long)(avalid ? ar : 0) * K;

    // async staging assignment: thread t covers 32 contiguous bytes of sB
    const int  sc    = tid >> 1;              // column 0..63
    const int  shalf = tid & 1;               // 32-byte half of the column
    const int  gcol  = colB0 + sc;
    _Float16*  sdst  = sB + sc * 32 + shalf * 16;          // this thread's LDS dest
    const unsigned ldsAddr = (unsigned)(uintptr_t)sdst;    // LDS byte address (addr[31:0])
    const _Float16* gsrc = Bt + (long)((gcol < M) ? gcol : 0) * K + shalf * 16;

    v8f acc[4] = {};

    for (int k0 = 0; k0 < K; k0 += 32) {
        if (k0 + 32 <= K) {
            // ---- stage B tile into LDS (gfx1250 async-to-LDS path) ----
            __syncthreads();   // previous iteration finished reading sB
            {
                unsigned long long ga =
                    (unsigned long long)(uintptr_t)(gsrc + k0);
                asm volatile("global_load_async_to_lds_b128 %0, %1, off"
                             :: "v"(ldsAddr), "v"(ga) : "memory");
                asm volatile("global_load_async_to_lds_b128 %0, %1, off offset:16"
                             :: "v"(ldsAddr), "v"(ga) : "memory");
            }
            asm volatile("s_wait_asynccnt 0x0" ::: "memory");
            __syncthreads();   // whole tile visible to all waves

            // ---- A fragment (global, per wave) ----
            if (avalid && k0 + 64 <= K)
                __builtin_prefetch(arow + k0 + 32, 0, 1);
            v16h a;
            if (avalid) {
                // A 16-bit layout: VGPR0-3: K = kh*8+0..7 ; VGPR4-7: K = 16+kh*8+0..7
                const v8h a0 = *reinterpret_cast<const v8h*>(arow + k0 + kh * 8);
                const v8h a1 = *reinterpret_cast<const v8h*>(arow + k0 + 16 + kh * 8);
#pragma unroll
                for (int e = 0; e < 8; ++e) { a[e] = a0[e]; a[e + 8] = a1[e]; }
            } else {
#pragma unroll
                for (int e = 0; e < 16; ++e) a[e] = (_Float16)0.0f;
            }

            // ---- 4 B fragments from LDS (ds_load_b128), 4 WMMAs ----
#pragma unroll
            for (int j = 0; j < 4; ++j) {
                const int  lc     = j * 16 + mr;          // local col in block tile
                const bool bvalid = (colB0 + lc) < M;
                v16h b;
                // B 16-bit layout: elems 0-7: K = kh*8+0..7 ; elems 8-15: K = 16+kh*8+0..7
                const v8h b0 = *reinterpret_cast<const v8h*>(sB + lc * 32 + kh * 8);
                const v8h b1 = *reinterpret_cast<const v8h*>(sB + lc * 32 + 16 + kh * 8);
#pragma unroll
                for (int e = 0; e < 8; ++e) { b[e] = b0[e]; b[e + 8] = b1[e]; }
                if (!bvalid) {
#pragma unroll
                    for (int e = 0; e < 16; ++e) b[e] = (_Float16)0.0f;
                }
                acc[j] = __builtin_amdgcn_wmma_f32_16x16x32_f16(
                    false, a, false, b, (short)0, acc[j], false, false);
            }
        } else {
            // ---- K tail: scalar gather with zero padding (no LDS) ----
            v16h a;
#pragma unroll
            for (int e = 0; e < 16; ++e) {
                int ka = k0 + ((e < 8) ? (kh * 8 + e) : (16 + kh * 8 + (e - 8)));
                a[e] = (avalid && ka < K) ? arow[ka] : (_Float16)0.0f;
            }
#pragma unroll
            for (int j = 0; j < 4; ++j) {
                const int  gc = colB0 + j * 16 + mr;
                const _Float16* brow = Bt + (long)((gc < M) ? gc : 0) * K;
                v16h b;
#pragma unroll
                for (int e = 0; e < 16; ++e) {
                    int kb = k0 + ((e < 8) ? (kh * 8 + e) : (16 + kh * 8 + (e - 8)));
                    b[e] = (gc < M && kb < K) ? brow[kb] : (_Float16)0.0f;
                }
                acc[j] = __builtin_amdgcn_wmma_f32_16x16x32_f16(
                    false, a, false, b, (short)0, acc[j], false, false);
            }
        }
    }

    // C/D layout: lanes 0-15 -> M=v, lanes 16-31 -> M=v+8; N = lane&15
#pragma unroll
    for (int j = 0; j < 4; ++j) {
#pragma unroll
        for (int v = 0; v < 8; ++v) {
            int r = row0 + v + kh * 8;
            int c = colB0 + j * 16 + mr;
            if (r < N && c < M) C[(long)r * M + c] = acc[j][v] + bias[c];
        }
    }
}

// ------------------------------- edge phase --------------------------------

__global__ void k_gat_alpha(const float* __restrict__ xl, const float* __restrict__ xr,
                            const int* __restrict__ src, const int* __restrict__ dst,
                            const float* __restrict__ att, float* __restrict__ alpha,
                            float* __restrict__ nmax, long E, int H, int C)
{
    long t = blockIdx.x * (long)blockDim.x + threadIdx.x;
    if (t >= E * (long)H) return;
    long e = t / H; int h = (int)(t % H);
    int s = src[e], d = dst[e];
    const float* pl = xl + ((long)s * H + h) * C;
    const float* pr = xr + ((long)d * H + h) * C;
    const float* pa = att + (long)h * C;
    float acc = 0.0f;
    for (int c = 0; c < C; ++c) {
        float z = pl[c] + pr[c];
        z = z > 0.0f ? z : 0.2f * z;     // leaky relu 0.2
        acc += z * pa[c];
    }
    alpha[t] = acc;
    atomicMaxF(&nmax[(long)d * H + h], acc);
}

__global__ void k_trans_alpha(const float* __restrict__ q, const float* __restrict__ kk,
                              const int* __restrict__ src, const int* __restrict__ dst,
                              float* __restrict__ alpha, float* __restrict__ nmax,
                              long E, int H, int C, float scale)
{
    long t = blockIdx.x * (long)blockDim.x + threadIdx.x;
    if (t >= E * (long)H) return;
    long e = t / H; int h = (int)(t % H);
    int s = src[e], d = dst[e];
    const float* pq = q  + ((long)d * H + h) * C;
    const float* pk = kk + ((long)s * H + h) * C;
    float acc = 0.0f;
    for (int c = 0; c < C; ++c) acc += pq[c] * pk[c];
    acc *= scale;
    alpha[t] = acc;
    atomicMaxF(&nmax[(long)d * H + h], acc);
}

__global__ void k_segexp(float* __restrict__ alpha, const float* __restrict__ nmax,
                         const int* __restrict__ dst, float* __restrict__ nsum,
                         long E, int H)
{
    long t = blockIdx.x * (long)blockDim.x + threadIdx.x;
    if (t >= E * (long)H) return;
    long e = t / H; int h = (int)(t % H);
    int d = dst[e];
    float m = nmax[(long)d * H + h];
    if (!(m * 0.0f == 0.0f)) m = 0.0f;   // replace +/-inf, nan with 0
    float ex = expf(alpha[t] - m);
    alpha[t] = ex;
    atomicAdd(&nsum[(long)d * H + h], ex);
}

__global__ void k_agg(const float* __restrict__ feat, const float* __restrict__ alpha,
                      const float* __restrict__ nsum, const int* __restrict__ src,
                      const int* __restrict__ dst, float* __restrict__ out,
                      long E, int H, int C)
{
    long t = blockIdx.x * (long)blockDim.x + threadIdx.x;
    if (t >= E * (long)H) return;
    long e = t / H; int h = (int)(t % H);
    int s = src[e], d = dst[e];
    float w = alpha[t] / (nsum[(long)d * H + h] + 1e-16f);
    const float* pf = feat + ((long)s * H + h) * C;
    float*       po = out  + ((long)d * H + h) * C;
    for (int c = 0; c < C; ++c) atomicAdd(&po[c], pf[c] * w);
}

__global__ void k_gat_finalize(const float* __restrict__ acc, const float* __restrict__ bias,
                               float* __restrict__ out, int N, int H, int C, int concat)
{
    long t = blockIdx.x * (long)blockDim.x + threadIdx.x;
    if (concat) {
        long n = (long)N * H * C; if (t >= n) return;
        int col = (int)(t % ((long)H * C));
        out[t] = acc[t] + bias[col];
    } else {
        long n = (long)N * C; if (t >= n) return;
        long node = t / C; int c = (int)(t % C);
        float s = 0.0f;
        for (int h = 0; h < H; ++h) s += acc[(node * H + h) * C + c];
        out[t] = s / (float)H + bias[c];
    }
}

// ------------------------------- graph norm --------------------------------

__global__ void k_colsum(const float* __restrict__ x, float* __restrict__ sums, int N, int D) {
    long t = blockIdx.x * (long)blockDim.x + threadIdx.x;
    if (t >= (long)N * D) return;
    atomicAdd(&sums[(int)(t % D)], x[t]);
}

__global__ void k_center(float* __restrict__ x, const float* __restrict__ sums,
                         const float* __restrict__ ms, float* __restrict__ vars, int N, int D) {
    long t = blockIdx.x * (long)blockDim.x + threadIdx.x;
    if (t >= (long)N * D) return;
    int c = (int)(t % D);
    float o = x[t] - (sums[c] / (float)N) * ms[c];
    x[t] = o;
    atomicAdd(&vars[c], o * o);
}

__global__ void k_gnorm_final(float* __restrict__ x, const float* __restrict__ vars,
                              const float* __restrict__ wt, const float* __restrict__ bs,
                              int N, int D, int do_gelu) {
    long t = blockIdx.x * (long)blockDim.x + threadIdx.x;
    if (t >= (long)N * D) return;
    int c = (int)(t % D);
    float inv = rsqrtf(vars[c] / (float)N + 1e-5f);
    float z = x[t] * inv * wt[c] + bs[c];
    x[t] = do_gelu ? gelu_f(z) : z;
}

// ------------------------------ pooling + head -----------------------------

__global__ void k_pool(const float* __restrict__ x, const int* __restrict__ batch,
                       float* __restrict__ gsum, float* __restrict__ gmax,
                       float* __restrict__ gcnt, int N, int D) {
    long t = blockIdx.x * (long)blockDim.x + threadIdx.x;
    if (t >= (long)N * D) return;
    long n = t / D; int c = (int)(t % D);
    int g = batch[n];
    atomicAdd(&gsum[(long)g * D + c], x[t]);
    atomicMaxF(&gmax[(long)g * D + c], x[t]);
    if (c == 0) atomicAdd(&gcnt[g], 1.0f);
}

__global__ void k_head(const float* __restrict__ gsum, const float* __restrict__ gmax,
                       const float* __restrict__ gcnt, const float* __restrict__ W,
                       const float* __restrict__ wb, const float* __restrict__ lnw,
                       const float* __restrict__ lnb, float* __restrict__ out, int G, int D) {
    int g = threadIdx.x;
    if (g >= G) return;
    float cnt = fmaxf(gcnt[g], 1.0f);
    float y[18];
    for (int o = 0; o < D; ++o) {
        float a = wb[o], b = wb[o];
        for (int i = 0; i < D; ++i) {
            float w = W[i * D + o];
            a += (gsum[(long)g * D + i] / cnt) * w;
            b += gmax[(long)g * D + i] * w;
        }
        y[o] = a + b;
    }
    float m = 0.0f;
    for (int o = 0; o < D; ++o) m += y[o];
    m /= (float)D;
    float v = 0.0f;
    for (int o = 0; o < D; ++o) { float d = y[o] - m; v += d * d; }
    v /= (float)D;
    float inv = rsqrtf(v + 1e-5f);
    for (int o = 0; o < D; ++o) {
        float z = (y[o] - m) * inv * lnw[o] + lnb[o];
        out[(long)g * D + o] = gelu_f(z);
    }
}

// ------------------------------- host helpers ------------------------------

namespace {

inline unsigned nblk(long n, int b) { return (unsigned)((n + b - 1) / b); }

struct Ws {
    float *cur, *nxt, *t0, *t1, *t2, *acc;
    _Float16 *hx, *hw;
    float *alpha, *nmax, *nsum, *sums, *vars, *gsum, *gmax, *gcnt;
    const int *src, *dst;
    int N; long E;
    hipStream_t s;
};

void conv_x16(const float* x, _Float16* hx, long n, hipStream_t s) {
    k_h16<<<nblk(n, 256), 256, 0, s>>>(x, hx, n);
}

void gemm(const Ws& w, const _Float16* hx, int K, const float* W, const float* b,
          int M, float* Y) {
    k_w16t<<<nblk((long)K * M, 256), 256, 0, w.s>>>(W, w.hw, K, M);
    dim3 grid((unsigned)((M + 63) / 64), (unsigned)((w.N + 63) / 64));
    k_wmma_gemm_bias<<<grid, 128, 0, w.s>>>(hx, w.hw, b, Y, w.N, K, M);
}

void softmax_agg(const Ws& w, const float* feat, int H, int C, float* accum) {
    long EH = w.E * H;
    k_segexp<<<nblk(EH, 256), 256, 0, w.s>>>(w.alpha, w.nmax, w.dst, w.nsum, w.E, H);
    k_agg<<<nblk(EH, 256), 256, 0, w.s>>>(feat, w.alpha, w.nsum, w.src, w.dst, accum, w.E, H, C);
}

void gat(const Ws& w, const _Float16* hx, int K,
         const float* lW, const float* lb, const float* rW, const float* rb,
         const float* att, const float* bias, int H, int C, int concat, float* out) {
    gemm(w, hx, K, lW, lb, H * C, w.t0);   // xl
    gemm(w, hx, K, rW, rb, H * C, w.t1);   // xr
    long NH = (long)w.N * H;
    k_fill<<<nblk(NH, 256), 256, 0, w.s>>>(w.nmax, -HUGE_VALF, NH);
    k_fill<<<nblk(NH, 256), 256, 0, w.s>>>(w.nsum, 0.0f, NH);
    k_fill<<<nblk((long)w.N * H * C, 256), 256, 0, w.s>>>(w.acc, 0.0f, (long)w.N * H * C);
    long EH = w.E * H;
    k_gat_alpha<<<nblk(EH, 256), 256, 0, w.s>>>(w.t0, w.t1, w.src, w.dst, att,
                                                w.alpha, w.nmax, w.E, H, C);
    softmax_agg(w, w.t0, H, C, w.acc);
    long on = concat ? (long)w.N * H * C : (long)w.N * C;
    k_gat_finalize<<<nblk(on, 256), 256, 0, w.s>>>(w.acc, bias, out, w.N, H, C, concat);
}

void trans(const Ws& w, const _Float16* hx, int K,
           const float* qW, const float* qb, const float* kW, const float* kb,
           const float* vW, const float* vb, const float* sW, const float* sb,
           int H, int C, float* out) {
    gemm(w, hx, K, qW, qb, H * C, w.t0);   // q
    gemm(w, hx, K, kW, kb, H * C, w.t1);   // k
    gemm(w, hx, K, vW, vb, H * C, w.t2);   // v
    gemm(w, hx, K, sW, sb, H * C, out);    // skip (aggregation adds into it)
    long NH = (long)w.N * H;
    k_fill<<<nblk(NH, 256), 256, 0, w.s>>>(w.nmax, -HUGE_VALF, NH);
    k_fill<<<nblk(NH, 256), 256, 0, w.s>>>(w.nsum, 0.0f, NH);
    long EH = w.E * H;
    k_trans_alpha<<<nblk(EH, 256), 256, 0, w.s>>>(w.t0, w.t1, w.src, w.dst, w.alpha,
                                                  w.nmax, w.E, H, C,
                                                  1.0f / sqrtf((float)C));
    softmax_agg(w, w.t2, H, C, out);
}

void gnorm(const Ws& w, float* x, int D, const float* ms, const float* wt,
           const float* bs, int do_gelu) {
    k_fill<<<nblk(D, 256), 256, 0, w.s>>>(w.sums, 0.0f, D);
    k_fill<<<nblk(D, 256), 256, 0, w.s>>>(w.vars, 0.0f, D);
    long ND = (long)w.N * D;
    k_colsum<<<nblk(ND, 256), 256, 0, w.s>>>(x, w.sums, w.N, D);
    k_center<<<nblk(ND, 256), 256, 0, w.s>>>(x, w.sums, ms, w.vars, w.N, D);
    k_gnorm_final<<<nblk(ND, 256), 256, 0, w.s>>>(x, w.vars, wt, bs, w.N, D, do_gelu);
}

} // namespace

// --------------------------------- launch ----------------------------------

extern "C" void kernel_launch(void* const* d_in, const int* in_sizes, int n_in,
                              void* d_out, int out_size, void* d_ws, size_t ws_size,
                              hipStream_t stream) {
    const float* x0   = (const float*)d_in[0];
    const int*  eidx  = (const int*)d_in[1];
    const int*  batch = (const int*)d_in[2];
    const int   N = in_sizes[0] / 9;
    const long  E = in_sizes[1] / 2;

    // Param leaves follow jax tree flatten (alphabetical dict keys).
    auto P = [&](int i) -> const float* {
        int j = (i < n_in) ? i : (n_in - 1);
        return (const float*)d_in[j];
    };
    int p = 3;
    // "conv" (GATv2 144 -> 8*36): att, bias, lin_l{W,b}, lin_r{W,b}
    const float *c_att = P(p++), *c_bias = P(p++), *c_lW = P(p++), *c_lb = P(p++),
                *c_rW = P(p++), *c_rb = P(p++);
    // "final_lin": W, b
    const float *fl_W = P(p++), *fl_b = P(p++);
    // "final_norm": bias, weight
    const float *fn_b = P(p++), *fn_w = P(p++);
    // "initial_conv" (GATv2 9 -> 8*18)
    const float *ic_att = P(p++), *ic_bias = P(p++), *ic_lW = P(p++), *ic_lb = P(p++),
                *ic_rW = P(p++), *ic_rb = P(p++);
    // "initial_norm": bias, mean_scale, weight
    const float *in_b = P(p++), *in_ms = P(p++), *in_w = P(p++);
    // "layers": 11 x { conv{att,bias,lW,lb,rW,rb}, norm{b,ms,w}, transformer{k,q,skip,v} }
    struct LayerP {
        const float *att, *bias, *lW, *lb, *rW, *rb;
        const float *nb, *nms, *nw;
        const float *kW, *kb, *qW, *qb, *sW, *sb, *vW, *vb;
    } L[11];
    for (int i = 0; i < 11; ++i) {
        L[i] = { P(p++), P(p++), P(p++), P(p++), P(p++), P(p++),
                 P(p++), P(p++), P(p++),
                 P(p++), P(p++), P(p++), P(p++), P(p++), P(p++), P(p++), P(p++) };
    }
    // "norm": bias, mean_scale, weight
    const float *n_b = P(p++), *n_ms = P(p++), *n_w = P(p++);
    // "transformer" (initial, 144): k{W,b}, q{W,b}, skip{W,b}, v{W,b}
    const float *t_kW = P(p++), *t_kb = P(p++), *t_qW = P(p++), *t_qb = P(p++),
                *t_sW = P(p++), *t_sb = P(p++), *t_vW = P(p++), *t_vb = P(p++);

    // ---- workspace carve (~156 MB) ----
    char* base = (char*)d_ws;
    size_t off = 0;
    auto carve = [&](size_t bytes) -> void* {
        void* r = base + off;
        off = (off + bytes + 255) & ~(size_t)255;
        return r;
    };
    const size_t nodeBuf = (size_t)N * 288 * sizeof(float);
    Ws w;
    w.cur = (float*)carve(nodeBuf);
    w.nxt = (float*)carve(nodeBuf);
    w.t0  = (float*)carve(nodeBuf);
    w.t1  = (float*)carve(nodeBuf);
    w.t2  = (float*)carve(nodeBuf);
    w.acc = (float*)carve(nodeBuf);
    w.hx  = (_Float16*)carve((size_t)N * 288 * sizeof(_Float16));
    w.hw  = (_Float16*)carve((size_t)288 * 288 * sizeof(_Float16));
    w.alpha = (float*)carve((size_t)E * 8 * sizeof(float));
    w.nmax  = (float*)carve((size_t)N * 8 * sizeof(float));
    w.nsum  = (float*)carve((size_t)N * 8 * sizeof(float));
    w.sums  = (float*)carve(288 * sizeof(float));
    w.vars  = (float*)carve(288 * sizeof(float));
    w.gsum  = (float*)carve(64 * 18 * sizeof(float));
    w.gmax  = (float*)carve(64 * 18 * sizeof(float));
    w.gcnt  = (float*)carve(64 * sizeof(float));
    w.src = eidx;
    w.dst = eidx + E;
    w.N = N; w.E = E; w.s = stream;

    // ---- forward pass ----
    // initial GATv2 (9 -> 144) + graphnorm + gelu
    conv_x16(x0, w.hx, (long)N * 9, stream);
    gat(w, w.hx, 9, ic_lW, ic_lb, ic_rW, ic_rb, ic_att, ic_bias, 8, 18, 1, w.cur);
    gnorm(w, w.cur, 144, in_ms, in_w, in_b, 1);

    // initial transformer (144 -> 144)
    conv_x16(w.cur, w.hx, (long)N * 144, stream);
    trans(w, w.hx, 144, t_qW, t_qb, t_kW, t_kb, t_vW, t_vb, t_sW, t_sb, 8, 18, w.nxt);

    // conv GATv2 (144 -> 288) + graphnorm + gelu
    conv_x16(w.nxt, w.hx, (long)N * 144, stream);
    gat(w, w.hx, 144, c_lW, c_lb, c_rW, c_rb, c_att, c_bias, 8, 36, 1, w.cur);
    gnorm(w, w.cur, 288, n_ms, n_w, n_b, 1);

    // 11 layers
    for (int i = 0; i < 11; ++i) {
        const int oc  = (i < 10) ? 36 : 18;
        const int h   = (i < 10) ? 8 : 1;
        const int cc  = (i < 10) ? 1 : 0;
        const int dim = cc ? oc * h : oc;

        conv_x16(w.cur, w.hx, (long)N * 288, stream);
        gat(w, w.hx, 288, L[i].lW, L[i].lb, L[i].rW, L[i].rb,
            L[i].att, L[i].bias, h, oc, cc, w.nxt);
        gnorm(w, w.nxt, dim, L[i].nms, L[i].nw, L[i].nb, 1);

        conv_x16(w.nxt, w.hx, (long)N * dim, stream);
        trans(w, w.hx, dim, L[i].qW, L[i].qb, L[i].kW, L[i].kb,
              L[i].vW, L[i].vb, L[i].sW, L[i].sb, h, oc, w.acc);

        if (dim == 288) {
            k_add_inplace<<<nblk((long)N * 288, 256), 256, 0, stream>>>(
                w.cur, w.acc, (long)N * 288);                 // residual
        } else {
            k_copy<<<nblk((long)N * dim, 256), 256, 0, stream>>>(
                w.cur, w.acc, (long)N * dim);
        }
    }

    // pooling over graphs + final linear + layernorm + gelu
    k_fill<<<nblk(64 * 18, 256), 256, 0, stream>>>(w.gsum, 0.0f, 64 * 18);
    k_fill<<<nblk(64 * 18, 256), 256, 0, stream>>>(w.gmax, -HUGE_VALF, 64 * 18);
    k_fill<<<1, 64, 0, stream>>>(w.gcnt, 0.0f, 64);
    k_pool<<<nblk((long)N * 18, 256), 256, 0, stream>>>(w.cur, batch, w.gsum, w.gmax,
                                                        w.gcnt, N, 18);
    k_head<<<1, 64, 0, stream>>>(w.gsum, w.gmax, w.gcnt, fl_W, fl_b, fn_w, fn_b,
                                 (float*)d_out, 64, 18);
}
